// _LSTMSingleLayer_37950331027586
// MI455X (gfx1250) — matrix operand order repaired and checked by reading, not compiled, not run on hardware
//
#include <hip/hip_runtime.h>
#include <hip/hip_bf16.h>

#define SEQ     2048
#define BATCH   32
#define IN_DIM  512
#define HID     512
#define GATES   (4*HID)           // 2048
#define NWG     32                // persistent workgroups (one per WGP)
#define CPW     16                // hidden columns per workgroup (HID/NWG)
#define NROWS   64                // gate rows staged per WGP (4 gates * CPW)
#define LDK     520               // padded K stride (halves) to dodge bank conflicts

typedef __attribute__((ext_vector_type(16))) _Float16 v16h;
typedef __attribute__((ext_vector_type(8)))  _Float16 v8h;
typedef __attribute__((ext_vector_type(8)))  float    v8f;

union FragU { v16h v; v8h h[2]; };

__device__ __forceinline__ float sigmoidf_fast(float v) {
    return 1.0f / (1.0f + __expf(-v));
}

// ---------------------------------------------------------------------------
// Prep: W_ih/W_hh -> f16, bias = b_ih+b_hh, h0 -> f16 buffer 0, ctr = 0
// ---------------------------------------------------------------------------
__global__ void lstm_prep(const float* __restrict__ Wih, const float* __restrict__ Whh,
                          const float* __restrict__ bih, const float* __restrict__ bhh,
                          const float* __restrict__ h0,
                          _Float16* __restrict__ gWih, _Float16* __restrict__ gWhh,
                          float* __restrict__ bias, _Float16* __restrict__ hbuf,
                          unsigned* __restrict__ ctr) {
    const size_t stride = (size_t)gridDim.x * blockDim.x;
    const size_t i0 = (size_t)blockIdx.x * blockDim.x + threadIdx.x;
    const size_t nW = (size_t)GATES * IN_DIM;            // 1,048,576
    for (size_t i = i0; i < nW; i += stride) {
        gWih[i] = (_Float16)Wih[i];
        gWhh[i] = (_Float16)Whh[i];
    }
    if (i0 < (size_t)GATES)       bias[i0] = bih[i0] + bhh[i0];
    if (i0 < (size_t)BATCH*HID)   hbuf[i0] = (_Float16)h0[i0];
    if (i0 == 0)                  *ctr = 0u;
}

// ---------------------------------------------------------------------------
// Persistent recurrent kernel: NWG blocks x 256 threads (8 waves).
// Block w owns hidden columns [w*16, w*16+16): 64 gate columns = 4 N-tiles,
// one per gate. Waves (mt,nt): mt in {0,1} batch halves, nt in {0..3} gates.
// ---------------------------------------------------------------------------
__global__ void __launch_bounds__(256, 1)
lstm_persistent(const float* __restrict__ x,        // [SEQ, 32, 512]
                const float* __restrict__ c0,       // [32, 512]
                const float* __restrict__ bias,     // [2048]
                const _Float16* __restrict__ gWih,  // [2048, 512] f16
                const _Float16* __restrict__ gWhh,  // [2048, 512] f16
                _Float16* __restrict__ hbuf,        // [2][32*512] f16 (double buffer)
                unsigned* __restrict__ ctr,
                float* __restrict__ out)            // hs | h_T | c_T
{
    __shared__ _Float16 sWih[NROWS * LDK];   // 66.5 KB : B slice of W_ih^T
    __shared__ _Float16 sWhh[NROWS * LDK];   // 66.5 KB : B slice of W_hh^T
    __shared__ _Float16 sX[BATCH * LDK];     // 33.3 KB : A = x_t (f16)
    __shared__ _Float16 sH[BATCH * LDK];     // 33.3 KB : A = h   (f16)
    __shared__ float    sGates[4][BATCH][CPW]; // 8 KB : per-gate pre-activation

    const int w    = blockIdx.x;             // 0..31
    const int tid  = threadIdx.x;            // 0..255
    const int lane = tid & 31;
    const int wid  = tid >> 5;               // 0..7
    const int mt   = wid >> 2;               // 0..1 (batch row tile)
    const int nt   = wid & 3;                // 0..3 == gate index

    // ---- stage weight slices into LDS (once), [n][k] major, n = gate*16+col
    for (int q = tid; q < NROWS * (IN_DIM/8); q += 256) {
        int n  = q >> 6;                      // 0..63
        int ch = q & 63;                      // 8-half chunk within row
        int grow = (n >> 4) * HID + w * CPW + (n & 15);   // global gate row
        *(uint4*)(sWih + n * LDK + ch * 8) =
            *(const uint4*)(gWih + (size_t)grow * IN_DIM + ch * 8);
        *(uint4*)(sWhh + n * LDK + ch * 8) =
            *(const uint4*)(gWhh + (size_t)grow * HID + ch * 8);
    }

    // ---- per-thread elementwise assignment: 2 (row,col) cells per thread
    const int e0 = tid * 2, e1 = tid * 2 + 1;
    const int r0 = e0 >> 4, cl0 = e0 & 15;
    const int r1 = e1 >> 4, cl1 = e1 & 15;
    const int hc0 = w * CPW + cl0;
    const int hc1 = w * CPW + cl1;
    float cst0 = c0[r0 * HID + hc0];
    float cst1 = c0[r1 * HID + hc1];
    const float bI0 = bias[0*HID + hc0], bF0 = bias[1*HID + hc0];
    const float bG0 = bias[2*HID + hc0], bO0 = bias[3*HID + hc0];
    const float bI1 = bias[0*HID + hc1], bF1 = bias[1*HID + hc1];
    const float bG1 = bias[2*HID + hc1], bO1 = bias[3*HID + hc1];

    // fragment addressing (per ISA VGPR layouts, wave32)
    const int hi   = lane >> 4;               // half-wave select
    const int arow = mt * 16 + (lane & 15);   // A: batch row
    const int brow = nt * 16 + (lane & 15);   // B: local gate row (= W row)

    for (int t = 0; t < SEQ; ++t) {
        // ---- load x_t (f32 -> f16) into sX
        const float* xt = x + (size_t)t * BATCH * IN_DIM;
        for (int q = tid; q < BATCH * IN_DIM / 4; q += 256) {
            float4 v = ((const float4*)xt)[q];
            int row = q >> 7;                 // (q*4)/512
            int k   = (q * 4) & 511;
            _Float16* d = sX + row * LDK + k;
            d[0] = (_Float16)v.x; d[1] = (_Float16)v.y;
            d[2] = (_Float16)v.z; d[3] = (_Float16)v.w;
        }
        // ---- load h (f16) into sH from double buffer (t&1)
        const _Float16* hsrc = hbuf + (size_t)(t & 1) * (BATCH * HID);
        for (int q = tid; q < BATCH * HID / 8; q += 256) {
            int row = q >> 6;                 // (q*8)/512
            int k   = (q * 8) & 511;
            *(uint4*)(sH + row * LDK + k) = ((const uint4*)hsrc)[q];
        }
        __syncthreads();

        // ---- gates tile: acc = x_t @ Wih^T + h @ Whh^T  (K = 512 + 512)
        v8f acc = {};
        #pragma unroll
        for (int kk = 0; kk < IN_DIM / 32; ++kk) {
            const int k0 = kk * 32;
            FragU a, b;
            a.h[0] = *(const v8h*)(sX + arow * LDK + k0 + hi * 8);
            a.h[1] = *(const v8h*)(sX + arow * LDK + k0 + 16 + hi * 8);
            b.h[0] = *(const v8h*)(sWih + brow * LDK + k0 + hi * 8);
            b.h[1] = *(const v8h*)(sWih + brow * LDK + k0 + 16 + hi * 8);
            acc = __builtin_amdgcn_wmma_f32_16x16x32_f16(
                      false, a.v, false, b.v, (short)0, acc, false, false);
        }
        #pragma unroll
        for (int kk = 0; kk < HID / 32; ++kk) {
            const int k0 = kk * 32;
            FragU a, b;
            a.h[0] = *(const v8h*)(sH + arow * LDK + k0 + hi * 8);
            a.h[1] = *(const v8h*)(sH + arow * LDK + k0 + 16 + hi * 8);
            b.h[0] = *(const v8h*)(sWhh + brow * LDK + k0 + hi * 8);
            b.h[1] = *(const v8h*)(sWhh + brow * LDK + k0 + 16 + hi * 8);
            acc = __builtin_amdgcn_wmma_f32_16x16x32_f16(
                      false, a.v, false, b.v, (short)0, acc, false, false);
        }

        // ---- spill C tile (this wave's gate) to LDS
        {
            const int Mb = hi * 8;            // lanes 16-31 hold M = 8..15
            const int N  = lane & 15;
            #pragma unroll
            for (int r = 0; r < 8; ++r)
                sGates[nt][mt * 16 + Mb + r][N] = acc[r];
        }
        __syncthreads();

        // ---- fused LSTM cell elementwise (2 cells/thread), write hs + h_f16
        float* outh = out + (size_t)t * BATCH * HID;
        _Float16* hdst = hbuf + (size_t)((t + 1) & 1) * (BATCH * HID);
        {
            float iv = sigmoidf_fast(sGates[0][r0][cl0] + bI0);
            float fv = sigmoidf_fast(sGates[1][r0][cl0] + bF0);
            float gv = tanhf(sGates[2][r0][cl0] + bG0);
            float ov = sigmoidf_fast(sGates[3][r0][cl0] + bO0);
            cst0 = fv * cst0 + iv * gv;
            float hv = ov * tanhf(cst0);
            outh[r0 * HID + hc0] = hv;
            hdst[r0 * HID + hc0] = (_Float16)hv;
            if (t == SEQ - 1) {
                out[(size_t)SEQ*BATCH*HID + r0*HID + hc0] = hv;
                out[(size_t)SEQ*BATCH*HID + BATCH*HID + r0*HID + hc0] = cst0;
            }
        }
        {
            float iv = sigmoidf_fast(sGates[0][r1][cl1] + bI1);
            float fv = sigmoidf_fast(sGates[1][r1][cl1] + bF1);
            float gv = tanhf(sGates[2][r1][cl1] + bG1);
            float ov = sigmoidf_fast(sGates[3][r1][cl1] + bO1);
            cst1 = fv * cst1 + iv * gv;
            float hv = ov * tanhf(cst1);
            outh[r1 * HID + hc1] = hv;
            hdst[r1 * HID + hc1] = (_Float16)hv;
            if (t == SEQ - 1) {
                out[(size_t)SEQ*BATCH*HID + r1*HID + hc1] = hv;
                out[(size_t)SEQ*BATCH*HID + BATCH*HID + r1*HID + hc1] = cst1;
            }
        }

        // ---- grid barrier across the 32 persistent workgroups
        __threadfence();
        __syncthreads();
        if (tid == 0) {
            __hip_atomic_fetch_add(ctr, 1u, __ATOMIC_RELEASE, __HIP_MEMORY_SCOPE_AGENT);
            const unsigned target = (unsigned)(t + 1) * NWG;
            while (__hip_atomic_load(ctr, __ATOMIC_ACQUIRE, __HIP_MEMORY_SCOPE_AGENT) < target)
                __builtin_amdgcn_s_sleep(1);
        }
        __syncthreads();
        __threadfence();
    }
}

// ---------------------------------------------------------------------------
extern "C" void kernel_launch(void* const* d_in, const int* in_sizes, int n_in,
                              void* d_out, int out_size, void* d_ws, size_t ws_size,
                              hipStream_t stream) {
    const float* x   = (const float*)d_in[0];
    const float* h0  = (const float*)d_in[1];
    const float* c0  = (const float*)d_in[2];
    const float* Wih = (const float*)d_in[3];
    const float* Whh = (const float*)d_in[4];
    const float* bih = (const float*)d_in[5];
    const float* bhh = (const float*)d_in[6];
    float* out = (float*)d_out;

    char* ws = (char*)d_ws;
    const size_t szW = (size_t)GATES * IN_DIM * sizeof(_Float16);  // 2 MB
    _Float16* gWih = (_Float16*)(ws);
    _Float16* gWhh = (_Float16*)(ws + szW);
    float*    bias = (float*)   (ws + 2*szW);                      // 8 KB
    _Float16* hbuf = (_Float16*)(ws + 2*szW + 16384);              // 2 x 32 KB
    unsigned* ctr  = (unsigned*)(ws + 2*szW + 16384 + 2*(size_t)BATCH*HID*sizeof(_Float16));

    lstm_prep<<<1024, 256, 0, stream>>>(Wih, Whh, bih, bhh, h0,
                                        gWih, gWhh, bias, hbuf, ctr);
    lstm_persistent<<<NWG, 256, 0, stream>>>(x, c0, bias, gWih, gWhh, hbuf, ctr, out);
}
